// Encoder_38259568673421
// MI455X (gfx1250) — compile-verified, matrix operand order checked by reading, not run on hardware
//
#include <hip/hip_runtime.h>

// ---------------------------------------------------------------------------
// GRU encoder for MI455X (gfx1250, wave32, WMMA + TDM).
//   Phase 1: gather + bf16 packs (w_ih, w_hh)
//   Phase 2: x_proj GEMM (T=4096,K=1024,N=6144), v_wmma_f32_16x16x32_bf16,
//            2-deep register pipeline held in place with unroll(1) +
//            sched_barrier so waits are partial (loadcnt<=10), not drains.
//   Phase 3: persistent cooperative scan: 64 WGs; each WG's 192KB w_hh slice
//            is DMA'd into LDS by the Tensor Data Mover (3 2D descriptors),
//            fenced with s_wait_tensorcnt; device-scope barrier per step.
// ---------------------------------------------------------------------------

typedef __bf16 bf16_t;
typedef __attribute__((ext_vector_type(16))) __bf16 v16bf;
typedef __attribute__((ext_vector_type(8)))  __bf16 v8bf;
typedef __attribute__((ext_vector_type(4)))  __bf16 v4bf;
typedef __attribute__((ext_vector_type(8)))  float  v8f;
typedef __attribute__((ext_vector_type(4)))  float  v4f;
typedef __attribute__((ext_vector_type(4)))  unsigned int u32x4;
typedef __attribute__((ext_vector_type(8)))  int i32x8;
typedef __attribute__((ext_vector_type(4)))  int i32x4;

#define T_LEN 4096
#define H_DIM 1024
#define G3    3072          // 3*H
#define NTOT  6144          // 2*3*H
#define NWG_SCAN 64
#define SLOTS 32            // hidden units per scan block
#define OUT_HS ((long)T_LEN * 2 * H_DIM)

#if defined(__AMDGCN__) && __has_builtin(__builtin_amdgcn_tensor_load_to_lds) && \
    __has_builtin(__builtin_amdgcn_s_wait_tensorcnt)
#define HAVE_TDM 1
#else
#define HAVE_TDM 0
#endif

#if defined(__AMDGCN__) && __has_builtin(__builtin_amdgcn_sched_barrier)
#define SCHED_FENCE() __builtin_amdgcn_sched_barrier(0)
#else
#define SCHED_FENCE() ((void)0)
#endif

// ---------------- f32 -> bf16 pack ------------------------------------------
__global__ __launch_bounds__(256)
void k_pack_bf16(const float* __restrict__ src, bf16_t* __restrict__ dst, long n) {
  long i = ((long)blockIdx.x * 256 + threadIdx.x) * 8;
  if (i + 8 <= n) {
    v4f a = *(const v4f*)(src + i);
    v4f b = *(const v4f*)(src + i + 4);
    v8bf o;
    o[0] = (bf16_t)a[0]; o[1] = (bf16_t)a[1]; o[2] = (bf16_t)a[2]; o[3] = (bf16_t)a[3];
    o[4] = (bf16_t)b[0]; o[5] = (bf16_t)b[1]; o[6] = (bf16_t)b[2]; o[7] = (bf16_t)b[3];
    *(v8bf*)(dst + i) = o;
  }
}

// ---------------- embedding gather + bf16 -----------------------------------
__global__ __launch_bounds__(256)
void k_gather_emb(const int* __restrict__ x, const float* __restrict__ emb,
                  bf16_t* __restrict__ dst) {
  const int t = blockIdx.x;
  const long row = (long)x[t] * H_DIM;
  const int i = threadIdx.x * 4;
  v4f a = *(const v4f*)(emb + row + i);
  v4bf o;
  o[0] = (bf16_t)a[0]; o[1] = (bf16_t)a[1]; o[2] = (bf16_t)a[2]; o[3] = (bf16_t)a[3];
  *(v4bf*)(dst + (long)t * H_DIM + i) = o;
}

// ---------------- state init (graph-replay safe) ----------------------------
__global__ __launch_bounds__(256)
void k_init(float* __restrict__ hbuf, unsigned* __restrict__ ctr) {
  const int i = blockIdx.x * 256 + threadIdx.x;
  if (i < 4096) hbuf[i] = 0.0f;
  if (i == 0)   *ctr = 0u;
}

// ---------------- x_proj GEMM -----------------------------------------------
__device__ __forceinline__ v16bf ldfrag(const bf16_t* p) {
  v8bf lo = *(const v8bf*)p;
  v8bf hi = *(const v8bf*)(p + 16);
  return __builtin_shufflevector(lo, hi, 0,1,2,3,4,5,6,7,8,9,10,11,12,13,14,15);
}

struct Frags { v16bf a, b0, b1, b2, b3; };

__global__ __launch_bounds__(256)
void k_gemm_wmma(const bf16_t* __restrict__ A, const bf16_t* __restrict__ B,
                 const float* __restrict__ bias, float* __restrict__ C) {
  const int lane  = threadIdx.x & 31;
  const int wav   = threadIdx.x >> 5;
  const int job   = blockIdx.x * 8 + wav;       // 24576 wave-jobs
  const int mtile = job / 96;
  const int ngrp  = job - mtile * 96;
  const int n0    = ngrp * 64;
  const int ml    = lane & 15;
  const int kh    = (lane >> 4) << 3;           // 0 or 8

  const bf16_t* arow = A + (long)(mtile * 16 + ml) * H_DIM + kh;
  const bf16_t* brow = B + (long)(n0 + ml) * H_DIM + kh;
  const long bs = 16l * H_DIM;

  v8f acc0 = {}, acc1 = {}, acc2 = {}, acc3 = {};

  auto loadf = [&](int kc) {
    Frags f;
    f.a  = ldfrag(arow + kc);
    f.b0 = ldfrag(brow + kc);
    f.b1 = ldfrag(brow + bs + kc);
    f.b2 = ldfrag(brow + 2 * bs + kc);
    f.b3 = ldfrag(brow + 3 * bs + kc);
    return f;
  };

  Frags f0 = loadf(0);
#pragma unroll 1
  for (int kc = 0; kc < H_DIM; kc += 64) {
    Frags f1 = loadf(kc + 32);                  // next chunk in flight
    SCHED_FENCE();                              // loads issued before MACs below
    acc0 = __builtin_amdgcn_wmma_f32_16x16x32_bf16(false, f0.a, false, f0.b0, (short)0, acc0, false, false);
    acc1 = __builtin_amdgcn_wmma_f32_16x16x32_bf16(false, f0.a, false, f0.b1, (short)0, acc1, false, false);
    acc2 = __builtin_amdgcn_wmma_f32_16x16x32_bf16(false, f0.a, false, f0.b2, (short)0, acc2, false, false);
    acc3 = __builtin_amdgcn_wmma_f32_16x16x32_bf16(false, f0.a, false, f0.b3, (short)0, acc3, false, false);
    if (kc + 64 < H_DIM) f0 = loadf(kc + 64);   // refill while f1 computes
    SCHED_FENCE();
    acc0 = __builtin_amdgcn_wmma_f32_16x16x32_bf16(false, f1.a, false, f1.b0, (short)0, acc0, false, false);
    acc1 = __builtin_amdgcn_wmma_f32_16x16x32_bf16(false, f1.a, false, f1.b1, (short)0, acc1, false, false);
    acc2 = __builtin_amdgcn_wmma_f32_16x16x32_bf16(false, f1.a, false, f1.b2, (short)0, acc2, false, false);
    acc3 = __builtin_amdgcn_wmma_f32_16x16x32_bf16(false, f1.a, false, f1.b3, (short)0, acc3, false, false);
  }

  const int mrow0 = mtile * 16 + ((lane >> 4) << 3);
  float* crow = C + (long)mrow0 * NTOT + n0 + ml;
  v8f accs[4] = {acc0, acc1, acc2, acc3};
#pragma unroll
  for (int j = 0; j < 4; ++j) {
    const float bv = bias[n0 + j * 16 + ml];
#pragma unroll
    for (int v = 0; v < 8; ++v)
      crow[(long)v * NTOT + j * 16] = accs[j][v] + bv;
  }
}

// ---------------- persistent cooperative GRU scan ---------------------------
// 64 blocks x 256 threads. Block b: direction d=b>>5, units [jbase, jbase+32).
// LDS weight layout (TDM-written): [gate g][slot][1024 bf16] = 3 x 64KB.
__global__ __launch_bounds__(256)
void k_gru_scan(const float* __restrict__ xproj, const bf16_t* __restrict__ whhbf,
                const float* __restrict__ bhh, float* __restrict__ hbuf,
                unsigned* __restrict__ ctr, float* __restrict__ out) {
  extern __shared__ unsigned char smem[];
  unsigned* wlds = (unsigned*)smem;                        // 49152 dwords (192KB)
  float* sh_h = (float*)(smem + SLOTS * 3 * 512 * 4);      // 1024 f32
  float* sb   = sh_h + H_DIM;                              // 96 f32

  const int b = blockIdx.x;
  const int tid = threadIdx.x;
  const int d = b >> 5;
  const int jbase = (b & 31) * SLOTS;
  const int lane = tid & 31;
  const int wav  = tid >> 5;

#if HAVE_TDM
  // Tensor Data Mover: 3 x (32-row x 2KB) 2D tiles, one per gate.
  if (wav == 0) {
    const unsigned long long gbase = (unsigned long long)(uintptr_t)whhbf;
    const unsigned ldsbase = (unsigned)(uintptr_t)wlds;    // flat[31:0] == LDS offset
#pragma unroll
    for (int g = 0; g < 3; ++g) {
      const unsigned long long ga =
          gbase + ((unsigned long long)(d * G3 + g * H_DIM + jbase) << 11); // rows * 2048B
      u32x4 g0;
      g0[0] = 1u;                                           // count=1, no gather
      g0[1] = ldsbase + (unsigned)(g * 65536);              // lds_addr
      g0[2] = (unsigned)ga;                                 // global_addr[31:0]
      g0[3] = (unsigned)((ga >> 32) & 0x01FFFFFFu) | 0x80000000u; // addr[56:32], type=2
      // group1: data_size=2B; tensor_dim0=1024, dim1=32; tile 1024x32;
      //         tensor_dim0_stride=1024.
      i32x8 g1 = { 0x00010000, 0x04000000, 0x00200000, 0x04000000, 32, 1024, 0, 0 };
      i32x4 z4 = {0, 0, 0, 0};
#if defined(__clang_major__) && (__clang_major__ >= 23)
      i32x8 z8 = {0, 0, 0, 0, 0, 0, 0, 0};
      __builtin_amdgcn_tensor_load_to_lds(g0, g1, z4, z4, z8, 0);
#else
      __builtin_amdgcn_tensor_load_to_lds(g0, g1, z4, z4, 0);
#endif
    }
    __builtin_amdgcn_s_wait_tensorcnt(0);
  }
#else
  // Fallback: cooperative dword copy of the pre-packed bf16 slice.
  {
    const unsigned* wsrc = (const unsigned*)whhbf;
    for (int idx = tid; idx < SLOTS * 3 * 512; idx += 256) {
      const int g = idx >> 14;                 // /16384
      const int i = idx & 16383;
      wlds[idx] = wsrc[((long)(d * G3 + g * H_DIM + jbase) << 9) + i];
    }
  }
#endif
  if (tid < SLOTS * 3) {
    const int slot = tid / 3, g = tid - slot * 3;
    sb[tid] = bhh[d * G3 + g * H_DIM + jbase + slot];
  }
  __syncthreads();

  for (int t = 0; t < T_LEN; ++t) {
    const float* hrd = hbuf + ((t & 1) * 2048) + d * H_DIM;
    for (int i = tid; i < H_DIM; i += 256) sh_h[i] = hrd[i];
    if (tid == 0 && t + 1 < T_LEN)
      __builtin_prefetch(xproj + (long)(t + 1) * NTOT + d * G3 + jbase, 0, 0);
    __syncthreads();

    float* hwr = hbuf + (((t + 1) & 1) * 2048) + d * H_DIM;
#pragma unroll
    for (int s = 0; s < 4; ++s) {
      const int slot = wav * 4 + s;
      const int j = jbase + slot;
      const unsigned* wr = wlds + slot * 512;  // gate stride = 16384 dwords
      float ar = 0.f, az = 0.f, an = 0.f;
      for (int i = lane; i < 512; i += 32) {
        const unsigned pr = wr[i], pz = wr[16384 + i], pn = wr[32768 + i];
        const float2 hv = *(const float2*)(sh_h + 2 * i);
        ar = fmaf(__uint_as_float(pr << 16),         hv.x, ar);
        ar = fmaf(__uint_as_float(pr & 0xFFFF0000u), hv.y, ar);
        az = fmaf(__uint_as_float(pz << 16),         hv.x, az);
        az = fmaf(__uint_as_float(pz & 0xFFFF0000u), hv.y, az);
        an = fmaf(__uint_as_float(pn << 16),         hv.x, an);
        an = fmaf(__uint_as_float(pn & 0xFFFF0000u), hv.y, an);
      }
#pragma unroll
      for (int off = 16; off > 0; off >>= 1) {
        ar += __shfl_xor(ar, off, 32);
        az += __shfl_xor(az, off, 32);
        an += __shfl_xor(an, off, 32);
      }
      if (lane == 0) {
        const float* xp = xproj + (long)t * NTOT + d * G3 + j;
        const float r = 1.0f / (1.0f + __expf(-(xp[0]     + ar + sb[slot * 3 + 0])));
        const float z = 1.0f / (1.0f + __expf(-(xp[H_DIM] + az + sb[slot * 3 + 1])));
        const float n = tanhf(xp[2 * H_DIM] + r * (an + sb[slot * 3 + 2]));
        const float hn = (1.0f - z) * n + z * sh_h[j];
        hwr[j] = hn;
        out[(long)t * 2048 + d * H_DIM + j] = hn;
      }
    }
    __threadfence();
    __syncthreads();
    if (tid == 0) {
      __hip_atomic_fetch_add(ctr, 1u, __ATOMIC_RELEASE, __HIP_MEMORY_SCOPE_AGENT);
      const unsigned target = (unsigned)NWG_SCAN * (unsigned)(t + 1);
      while (__hip_atomic_load(ctr, __ATOMIC_ACQUIRE, __HIP_MEMORY_SCOPE_AGENT) < target)
        __builtin_amdgcn_s_sleep(1);
    }
    __syncthreads();
  }

  if (d == 0 && tid < SLOTS) {
    const int j = jbase + tid;
    out[OUT_HS + j] = hbuf[j] + hbuf[H_DIM + j];
  }
}

// ---------------- host launcher ---------------------------------------------
extern "C" void kernel_launch(void* const* d_in, const int* in_sizes, int n_in,
                              void* d_out, int out_size, void* d_ws, size_t ws_size,
                              hipStream_t stream) {
  (void)in_sizes; (void)n_in; (void)out_size; (void)ws_size;
  const int*   x   = (const int*)d_in[0];
  const float* emb = (const float*)d_in[1];
  const float* wih = (const float*)d_in[2];
  const float* whh = (const float*)d_in[3];
  const float* bih = (const float*)d_in[4];
  const float* bhh = (const float*)d_in[5];
  float* out = (float*)d_out;
  char* ws = (char*)d_ws;

  bf16_t*   embbf = (bf16_t*)(ws);                    //   8,388,608 B
  bf16_t*   wihbf = (bf16_t*)(ws + 8388608);          //  12,582,912 B
  bf16_t*   whhbf = (bf16_t*)(ws + 20971520);         //  12,582,912 B
  float*    xproj = (float*) (ws + 33554432);         // 100,663,296 B
  float*    hbuf  = (float*) (ws + 134217728);        //      16,384 B
  unsigned* ctr   = (unsigned*)(ws + 134234112);      //           4 B

  const size_t scan_lds = (size_t)SLOTS * 3 * 512 * 4 + H_DIM * 4 + SLOTS * 3 * 4; // 201088
  (void)hipFuncSetAttribute((const void*)k_gru_scan,
                            hipFuncAttributeMaxDynamicSharedMemorySize, (int)scan_lds);

  k_pack_bf16<<<3072, 256, 0, stream>>>(wih, wihbf, (long)NTOT * H_DIM);
  k_pack_bf16<<<3072, 256, 0, stream>>>(whh, whhbf, (long)NTOT * H_DIM);
  k_gather_emb<<<T_LEN, 256, 0, stream>>>(x, emb, embbf);
  k_init<<<16, 256, 0, stream>>>(hbuf, ctr);

  k_gemm_wmma<<<3072, 256, 0, stream>>>(embbf, wihbf, bih, xproj);

  k_gru_scan<<<NWG_SCAN, 256, scan_lds, stream>>>(xproj, whhbf, bhh, hbuf, ctr, out);
}
